// MultiHeadAttention_12987981103491
// MI455X (gfx1250) — matrix operand order repaired and checked by reading, not compiled, not run on hardware
//
#include <hip/hip_runtime.h>

// ---------------------------------------------------------------------------
// MHA for MI455X (gfx1250, wave32, WMMA, async-to-LDS).
//   q,k,v:[2048,1024]f32  Wq/Wk/Wv:[16,1024,64]f32  Wo:[1024,1024]f32
//   out:[2048,1024]f32
// Pipeline (f16 WMMA, f32 accumulate):
//   1) proj_kernel   : qh[h][s][64], kh[h][s][64], vT[h][64][s]  (f16, ws)
//   2) flash_kernel  : flash attention -> head_cat[s][1024] (f16, ws)
//   3) outproj_kernel: head_cat @ Wo -> out (f32)
// ---------------------------------------------------------------------------

#define SEQ    2048
#define DMODEL 1024
#define DHEAD  64
#define HEADS  16

typedef __attribute__((ext_vector_type(16))) _Float16 v16h;
typedef __attribute__((ext_vector_type(8)))  _Float16 v8h;
typedef __attribute__((ext_vector_type(8)))  float    v8f;
typedef __attribute__((ext_vector_type(4)))  float    v4f;
typedef __attribute__((ext_vector_type(4)))  int      v4i;

// ---- async global->LDS staging (CDNA5), with portable fallback -------------
#if defined(__has_builtin)
#if __has_builtin(__builtin_amdgcn_global_load_async_to_lds_b128)
#define USE_ASYNC 1
#endif
#endif

#ifdef USE_ASYNC
// Probe-derived signature: (addrspace(1) v4i*, addrspace(3) v4i*, imm, imm)
typedef __attribute__((address_space(1))) v4i gas_v4i;
typedef __attribute__((address_space(3))) v4i las_v4i;
__device__ __forceinline__ void async_cp16(void* lds, const void* g) {
  __builtin_amdgcn_global_load_async_to_lds_b128((gas_v4i*)g, (las_v4i*)lds, 0, 0);
}
__device__ __forceinline__ void async_wait0() {
#if __has_builtin(__builtin_amdgcn_s_wait_asynccnt)
  __builtin_amdgcn_s_wait_asynccnt(0);
#else
  asm volatile("s_wait_asynccnt 0x0" ::: "memory");
#endif
}
#else
__device__ __forceinline__ void async_cp16(void* lds, const void* g) {
  *(v8h*)lds = *(const v8h*)g;  // 16B copy fallback
}
__device__ __forceinline__ void async_wait0() {}
#endif

__device__ __forceinline__ v8f wmma16x16x32(v16h a, v16h b, v8f c) {
  return __builtin_amdgcn_wmma_f32_16x16x32_f16(
      false, a, false, b, (short)0, c, false, false);
}

// A fragment (16x32 MxK, f16 source): lane L = row r=L&15, half i -> K =
// (i<8 ? i : i+8) + 8*(L>=16). Two contiguous 16B runs: [hb, hb+7], [hb+16, hb+23].
// ld in halves, multiple of 8; rows 16B-aligned.
__device__ __forceinline__ v16h load_A_frag_h(const _Float16* A, int ld, int lane) {
  const int r  = lane & 15;
  const int hb = (lane >> 4) << 3;
  const _Float16* p = A + r * ld + hb;
  v8h lo = *(const v8h*)p;
  v8h hi = *(const v8h*)(p + 16);
  v16h a;
#pragma unroll
  for (int i = 0; i < 8; ++i) { a[i] = lo[i]; a[8 + i] = hi[i]; }
  return a;
}

// Same fragment from an f32 tile (converted on gather). ld multiple of 4.
__device__ __forceinline__ v16h load_A_frag_f(const float* A, int ld, int lane) {
  const int r  = lane & 15;
  const int hb = (lane >> 4) << 3;
  const float* p = A + r * ld + hb;
  v4f x0 = *(const v4f*)(p);
  v4f x1 = *(const v4f*)(p + 4);
  v4f x2 = *(const v4f*)(p + 16);
  v4f x3 = *(const v4f*)(p + 20);
  v16h a;
#pragma unroll
  for (int i = 0; i < 4; ++i) {
    a[i]      = (_Float16)x0[i];
    a[4 + i]  = (_Float16)x1[i];
    a[8 + i]  = (_Float16)x2[i];
    a[12 + i] = (_Float16)x3[i];
  }
  return a;
}

// B fragment (32x16 KxN): lane L = col n=L&15, half i -> K = (i<8 ? i : i+8) + 8*(L>=16).
// Source holds element (k,n) at B[n*ld + k]; two 8-half runs per lane: [kb, kb+7], [kb+16, kb+23].
__device__ __forceinline__ v16h load_B_frag_h(const _Float16* B, size_t ld, int lane) {
  const int n  = lane & 15;
  const int kb = (lane >> 4) << 3;
  const _Float16* p = B + (size_t)n * ld + kb;
  v8h b0 = *(const v8h*)p;
  v8h b1 = *(const v8h*)(p + 16);
  v16h b;
#pragma unroll
  for (int i = 0; i < 8; ++i) { b[i] = b0[i]; b[8 + i] = b1[i]; }
  return b;
}

// xor-butterfly over 16-lane halves via ds_bpermute (wave32-safe).
__device__ __forceinline__ float lane_xor(float v, int m) {
  const int lane = (int)(threadIdx.x & 31u);
  int r = __builtin_amdgcn_ds_bpermute(((lane ^ m) << 2), __float_as_int(v));
  return __int_as_float(r);
}

// ---------------------------------------------------------------------------
// Kernel 1: QKV projections. grid=(SEQ/16, HEADS, 3), block=128 (4 waves).
// A tile staged f32 via async-to-LDS (converted on gather); B tile staged
// with float4 loads + f16 transpose scatter.
// ---------------------------------------------------------------------------
__global__ __launch_bounds__(128)
void proj_kernel(const float* __restrict__ q, const float* __restrict__ k,
                 const float* __restrict__ v, const float* __restrict__ Wq,
                 const float* __restrict__ Wk, const float* __restrict__ Wv,
                 _Float16* __restrict__ qh, _Float16* __restrict__ kh,
                 _Float16* __restrict__ vT) {
  const int s0 = blockIdx.x * 16;
  const int h  = blockIdx.y;
  const int z  = blockIdx.z;  // 0=q 1=k 2=v

  const float* X = (z == 0) ? q : (z == 1) ? k : v;
  const float* W = ((z == 0) ? Wq : (z == 1) ? Wk : Wv) +
                   (size_t)h * DMODEL * DHEAD;

  __shared__ __align__(16) float    Af[16 * 36];  // A tile f32, row stride 36
  __shared__ __align__(16) _Float16 Bl[64 * 40];  // (k,n) at Bl[n*40+k]

  const int tid  = threadIdx.x;
  const int lane = tid & 31;
  const int n0   = (tid >> 5) * 16;

  // Per-thread staging coordinates (exact division, no conditionals)
  const int ar  = tid >> 3;        // A: row 0..15
  const int ac4 = (tid & 7) * 4;   // A: col {0,4,...,28}

  v8f acc = {};

  for (int kt = 0; kt < DMODEL / 32; ++kt) {
    const int kb = kt * 32;
    // A tile: 512 f32 = 128 lanes x 16B, async DMA into LDS
    async_cp16(&Af[ar * 36 + ac4], X + (size_t)(s0 + ar) * DMODEL + kb + ac4);
    // B tile: 2048 f32 -> f16 transposed (vector loads, then scatter stores)
    v4f w[4];
#pragma unroll
    for (int p = 0; p < 4; ++p) {
      const int e  = tid + p * 128;      // float4 index 0..511
      const int kk = e >> 4;             // 0..31
      const int n4 = (e & 15) * 4;       // 0..60
      w[p] = *(const v4f*)(W + (size_t)(kb + kk) * DHEAD + n4);
    }
#pragma unroll
    for (int p = 0; p < 4; ++p) {
      const int e  = tid + p * 128;
      const int kk = e >> 4;
      const int n4 = (e & 15) * 4;
#pragma unroll
      for (int i = 0; i < 4; ++i) Bl[(n4 + i) * 40 + kk] = (_Float16)w[p][i];
    }
    async_wait0();
    __syncthreads();
    v16h a = load_A_frag_f(Af, 36, lane);
    v16h b = load_B_frag_h(Bl + n0 * 40, 40, lane);
    acc = wmma16x16x32(a, b, acc);
    __syncthreads();
  }

  const int n  = lane & 15;
  const int mb = (lane >> 4) * 8;
  if (z < 2) {
    _Float16* dst = ((z == 0) ? qh : kh) + ((size_t)h * SEQ + s0) * DHEAD;
#pragma unroll
    for (int j = 0; j < 8; ++j)
      dst[(size_t)(mb + j) * DHEAD + n0 + n] = (_Float16)acc[j];
  } else {  // V transposed: vT[h][dv][s]
    _Float16* dst = vT + (size_t)h * DHEAD * SEQ;
#pragma unroll
    for (int j = 0; j < 8; ++j)
      dst[(size_t)(n0 + n) * SEQ + s0 + mb + j] = (_Float16)acc[j];
  }
}

// ---------------------------------------------------------------------------
// Kernel 2: flash attention. grid=(SEQ/16, HEADS), block=32 (1 wave).
// Scores stay in accumulators (reference spills 256MB of [H,S,S] f32 > L2).
// ---------------------------------------------------------------------------
__global__ __launch_bounds__(32)
void flash_kernel(const _Float16* __restrict__ qh, const _Float16* __restrict__ kh,
                  const _Float16* __restrict__ vT, _Float16* __restrict__ hc) {
  const int s0   = blockIdx.x * 16;
  const int h    = blockIdx.y;
  const int lane = (int)(threadIdx.x & 31u);

  __shared__ __align__(16) _Float16 Pl[16 * 40];  // P tile, row stride 40

  const _Float16* Q  = qh + ((size_t)h * SEQ + s0) * DHEAD;
  const _Float16* Kh = kh + (size_t)h * SEQ * DHEAD;
  const _Float16* Vt = vT + (size_t)h * DHEAD * SEQ;

  // Q fragments with 1/sqrt(dk)=0.125 folded in (exact power of two in f16)
  v16h aq0 = load_A_frag_h(Q, DHEAD, lane);
  v16h aq1 = load_A_frag_h(Q + 32, DHEAD, lane);
#pragma unroll
  for (int i = 0; i < 16; ++i) {
    aq0[i] = aq0[i] * (_Float16)0.125f;
    aq1[i] = aq1[i] * (_Float16)0.125f;
  }

  float mrun[8], lrun[8];
#pragma unroll
  for (int j = 0; j < 8; ++j) { mrun[j] = -1e30f; lrun[j] = 0.0f; }
  v8f o0 = {}, o1 = {}, o2 = {}, o3 = {};

  const int n  = lane & 15;
  const int mb = (lane >> 4) * 8;

  for (int t = 0; t < SEQ / 32; ++t) {
    const _Float16* Kt = Kh + (size_t)(t * 32) * DHEAD;
    const _Float16* Vb = Vt + t * 32;

    if (t + 1 < SEQ / 32) {  // prefetch next K tile (4KB) and V tile rows
      __builtin_prefetch(Kt + 32 * DHEAD + lane * 64, 0, 3);
      __builtin_prefetch(Vt + (size_t)(lane * 2) * SEQ + t * 32 + 32, 0, 3);
    }

    // S tile [16 x 32]: two 16x16 D tiles, K-dim 64 (2 WMMAs each)
    v8f sa = {}, sb = {};
    sa = wmma16x16x32(aq0, load_B_frag_h(Kt, DHEAD, lane), sa);
    sa = wmma16x16x32(aq1, load_B_frag_h(Kt + 32, DHEAD, lane), sa);
    sb = wmma16x16x32(aq0, load_B_frag_h(Kt + 16 * DHEAD, DHEAD, lane), sb);
    sb = wmma16x16x32(aq1, load_B_frag_h(Kt + 16 * DHEAD + 32, DHEAD, lane), sb);

    // Online softmax (row stats via 16-lane bpermute butterflies)
    float al[8], p0[8], p1[8];
#pragma unroll
    for (int j = 0; j < 8; ++j) {
      float tm = fmaxf(sa[j], sb[j]);
#pragma unroll
      for (int m = 1; m < 16; m <<= 1) tm = fmaxf(tm, lane_xor(tm, m));
      const float mnew = fmaxf(mrun[j], tm);
      al[j] = __expf(mrun[j] - mnew);
      p0[j] = __expf(sa[j] - mnew);
      p1[j] = __expf(sb[j] - mnew);
      float rs = p0[j] + p1[j];
#pragma unroll
      for (int m = 1; m < 16; m <<= 1) rs += lane_xor(rs, m);
      lrun[j] = lrun[j] * al[j] + rs;
      mrun[j] = mnew;
    }
#pragma unroll
    for (int j = 0; j < 8; ++j) {
      o0[j] *= al[j]; o1[j] *= al[j]; o2[j] *= al[j]; o3[j] *= al[j];
    }

    // C-layout -> A-layout transpose of P through LDS
    __syncthreads();
#pragma unroll
    for (int j = 0; j < 8; ++j) {
      Pl[(mb + j) * 40 + n]      = (_Float16)p0[j];
      Pl[(mb + j) * 40 + 16 + n] = (_Float16)p1[j];
    }
    __syncthreads();
    const v16h ap = load_A_frag_h(Pl, 40, lane);

    // O += P @ V   (V transposed: element (k,dv) at Vt[dv*SEQ + k])
    o0 = wmma16x16x32(ap, load_B_frag_h(Vb + (size_t)0  * SEQ, SEQ, lane), o0);
    o1 = wmma16x16x32(ap, load_B_frag_h(Vb + (size_t)16 * SEQ, SEQ, lane), o1);
    o2 = wmma16x16x32(ap, load_B_frag_h(Vb + (size_t)32 * SEQ, SEQ, lane), o2);
    o3 = wmma16x16x32(ap, load_B_frag_h(Vb + (size_t)48 * SEQ, SEQ, lane), o3);
  }

  // Normalize, write head_cat[s][h*64+dv] (f16)
  _Float16* dst = hc + (size_t)s0 * (HEADS * DHEAD) + h * DHEAD;
#pragma unroll
  for (int j = 0; j < 8; ++j) {
    const float inv = 1.0f / lrun[j];
    const size_t row = (size_t)(mb + j) * (HEADS * DHEAD);
    dst[row + n]      = (_Float16)(o0[j] * inv);
    dst[row + 16 + n] = (_Float16)(o1[j] * inv);
    dst[row + 32 + n] = (_Float16)(o2[j] * inv);
    dst[row + 48 + n] = (_Float16)(o3[j] * inv);
  }
}

// ---------------------------------------------------------------------------
// Kernel 3: out = head_cat @ Wo. grid=(SEQ/16, 1024/64), block=128.
// A tile (f16) staged via async-to-LDS; B tile float4 + transpose scatter.
// ---------------------------------------------------------------------------
__global__ __launch_bounds__(128)
void outproj_kernel(const _Float16* __restrict__ hc, const float* __restrict__ Wo,
                    float* __restrict__ out) {
  const int s0  = blockIdx.x * 16;
  const int nb  = blockIdx.y * 64;
  const int tid = threadIdx.x;
  const int lane = tid & 31;
  const int n0   = (tid >> 5) * 16;

  __shared__ __align__(16) _Float16 Ah[16 * 40];
  __shared__ __align__(16) _Float16 Bl[64 * 40];

  const int ar  = tid >> 2;        // 0..31 (only 0..15 used)
  const int ac8 = (tid & 3) * 8;   // 0,8,16,24

  v8f acc = {};
  for (int kt = 0; kt < DMODEL / 32; ++kt) {
    const int kb = kt * 32;
    if (tid < 64)  // A tile: 16x32 f16 = 64 lanes x 16B
      async_cp16(&Ah[ar * 40 + ac8], hc + (size_t)(s0 + ar) * DMODEL + kb + ac8);
    v4f w[4];
#pragma unroll
    for (int p = 0; p < 4; ++p) {
      const int e  = tid + p * 128;
      const int kk = e >> 4;
      const int n4 = (e & 15) * 4;
      w[p] = *(const v4f*)(Wo + (size_t)(kb + kk) * DMODEL + nb + n4);
    }
#pragma unroll
    for (int p = 0; p < 4; ++p) {
      const int e  = tid + p * 128;
      const int kk = e >> 4;
      const int n4 = (e & 15) * 4;
#pragma unroll
      for (int i = 0; i < 4; ++i) Bl[(n4 + i) * 40 + kk] = (_Float16)w[p][i];
    }
    async_wait0();
    __syncthreads();
    v16h a = load_A_frag_h(Ah, 40, lane);
    v16h b = load_B_frag_h(Bl + n0 * 40, 40, lane);
    acc = wmma16x16x32(a, b, acc);
    __syncthreads();
  }

  const int nn = lane & 15;
  const int mb = (lane >> 4) * 8;
#pragma unroll
  for (int j = 0; j < 8; ++j)
    out[(size_t)(s0 + mb + j) * DMODEL + nb + n0 + nn] = acc[j];
}

// ---------------------------------------------------------------------------
extern "C" void kernel_launch(void* const* d_in, const int* in_sizes, int n_in,
                              void* d_out, int out_size, void* d_ws, size_t ws_size,
                              hipStream_t stream) {
  const float* q  = (const float*)d_in[0];
  const float* k  = (const float*)d_in[1];
  const float* v  = (const float*)d_in[2];
  const float* Wq = (const float*)d_in[3];
  const float* Wk = (const float*)d_in[4];
  const float* Wv = (const float*)d_in[5];
  const float* Wo = (const float*)d_in[6];
  float* out = (float*)d_out;

  const size_t HS = (size_t)HEADS * SEQ * DHEAD;
  _Float16* qh = (_Float16*)d_ws;
  _Float16* kh = qh + HS;
  _Float16* vT = kh + HS;
  _Float16* hc = vT + HS;  // 16 MB f16 scratch total

  proj_kernel<<<dim3(SEQ / 16, HEADS, 3), dim3(128), 0, stream>>>(
      q, k, v, Wq, Wk, Wv, qh, kh, vT);
  flash_kernel<<<dim3(SEQ / 16, HEADS), dim3(32), 0, stream>>>(qh, kh, vT, hc);
  outproj_kernel<<<dim3(SEQ / 16, DMODEL / 64), dim3(128), 0, stream>>>(hc, Wo, out);
}
